// TimeResampler_60189671686925
// MI455X (gfx1250) — matrix operand order repaired and checked by reading, not compiled, not run on hardware
//
#include <hip/hip_runtime.h>
#include <hip/hip_bf16.h>
#include <math.h>

// ---------------------------------------------------------------------------
// TimeResampler for MI455X (gfx1250, wave32).
// All GEMMs: v_wmma_f32_16x16x32_bf16, fp32->bf16 conversion during LDS
// staging, BM=64 x BN=256 x BK=32 tiles, double-buffered LDS (1 barrier per
// K-step, global loads for tile k+1 held in registers while WMMAs consume
// tile k).  LayerNorm + AdaLN modulation for the dominant kv projection
// (85% of the 1.5 TFLOP) is fused into the A-tile staging: per-row stats are
// precomputed (xp stats are layer-invariant), modulation folded into
// per-(batch,col) gain/bias, saving ~2.5 GB of HBM round-trips.
// Attention uses index-clamped (never predicated) gathers: padded score
// columns are overwritten with -inf and padded P columns are zero, so loads
// only need to be safe+finite, not masked.
// ---------------------------------------------------------------------------

typedef __attribute__((ext_vector_type(16))) __bf16 v16bf;
typedef __attribute__((ext_vector_type(8)))  float  v8f;

union BF16x16 { v16bf v; unsigned u[8]; __bf16 h[16]; };

static __device__ __forceinline__ unsigned pack_bf2(float a, float b) {
    union { __bf16 h[2]; unsigned u; } x;
    x.h[0] = (__bf16)a; x.h[1] = (__bf16)b;
    return x.u;
}

enum { EPI_NONE = 0, EPI_BIAS, EPI_BIAS_SILU, EPI_BIAS_TEMB, EPI_RESID, EPI_GELU };
enum { A_PLAIN = 0, A_LNCAT = 1 };

#define BM 64
#define BN 256
#define BK 32

// C[M,N] = A'[M,K] @ W[N,K]^T (+epilogue).  256 thr = 8 waves, each wave a
// 16x128 slab (8 WMMA tiles).  AMODE==A_LNCAT: A rows are built on the fly as
// rows 0..nx-1 of each rpbo-group from A (=X1) with LN(st1,g1,b1), remaining
// rows from X2 with LN(st2) and folded modulation gain/bias Gm/Bm[b, col].
template<int EPI, int AMODE>
__global__ __launch_bounds__(256) void gemm_bf16(
    const float* __restrict__ A, const float* __restrict__ W, float* __restrict__ C,
    int M, int N, int K,
    const float* __restrict__ aux1, const float* __restrict__ aux2, int rpb,
    const float* __restrict__ X2, const float2* __restrict__ st1,
    const float2* __restrict__ st2, const float* __restrict__ g1,
    const float* __restrict__ b1, const float* __restrict__ Gm,
    const float* __restrict__ Bm, int rpbo, int nx)
{
    __shared__ __bf16 lA[2][BM * BK];
    __shared__ __bf16 lB[2][BN * BK];

    const int t    = threadIdx.x;
    const int lane = t & 31;
    const int w    = t >> 5;
    const int wmg  = w >> 1;       // 0..3 : 16-row M tile
    const int wng  = w & 1;        // 0..1 : 128-col N half
    const int m0   = blockIdx.y * BM;
    const int n0   = blockIdx.x * BN;
    const int l16  = lane & 15;
    const int hl   = lane >> 4;
    const int arow = t >> 2, acol = (t & 3) * 8;   // A staging: 4 thr/row

    // resolve this thread's A-row source (fixed over the K loop)
    const float* asrc;
    float mu = 0.f, inv = 1.f;
    const float* gp = nullptr;
    const float* bp = nullptr;
    {
        const int r = m0 + arow;
        if (AMODE == A_LNCAT) {
            const int bi = r / rpbo, ri = r - bi * rpbo;
            if (ri < nx) {
                const int rr = bi * nx + ri;
                asrc = A + (size_t)rr * K;
                float2 s = st1[rr]; mu = s.x; inv = s.y;
                gp = g1; bp = b1;
            } else {
                const int rr = bi * 16 + (ri - nx);
                asrc = X2 + (size_t)rr * K;
                float2 s = st2[rr]; mu = s.x; inv = s.y;
                gp = Gm + (size_t)bi * K; bp = Bm + (size_t)bi * K;
            }
        } else {
            asrc = A + (size_t)r * K;
        }
    }
    const float* wsrc = W + (size_t)(n0 + t) * K;  // 1 W row per thread

    v8f acc[8];
#pragma unroll
    for (int i = 0; i < 8; ++i)
#pragma unroll
        for (int j = 0; j < 8; ++j) acc[i][j] = 0.f;

    float4 ra0, ra1, rg0, rg1, rc0, rc1, rw[8];

    auto loadA = [&](int kc) {
        const float4* sp = (const float4*)(asrc + kc + acol);
        ra0 = sp[0]; ra1 = sp[1];
        if (AMODE == A_LNCAT) {
            const float4* gq = (const float4*)(gp + kc + acol);
            const float4* bq = (const float4*)(bp + kc + acol);
            rg0 = gq[0]; rg1 = gq[1]; rc0 = bq[0]; rc1 = bq[1];
        }
    };
    auto loadW = [&](int kc) {
        const float4* sp = (const float4*)(wsrc + kc);
#pragma unroll
        for (int q = 0; q < 8; ++q) rw[q] = sp[q];
    };
    auto storeA = [&](int buf) {
        float4 a0 = ra0, a1 = ra1;
        if (AMODE == A_LNCAT) {
            a0.x = (a0.x - mu) * inv * rg0.x + rc0.x;
            a0.y = (a0.y - mu) * inv * rg0.y + rc0.y;
            a0.z = (a0.z - mu) * inv * rg0.z + rc0.z;
            a0.w = (a0.w - mu) * inv * rg0.w + rc0.w;
            a1.x = (a1.x - mu) * inv * rg1.x + rc1.x;
            a1.y = (a1.y - mu) * inv * rg1.y + rc1.y;
            a1.z = (a1.z - mu) * inv * rg1.z + rc1.z;
            a1.w = (a1.w - mu) * inv * rg1.w + rc1.w;
        }
        unsigned* d = (unsigned*)&lA[buf][arow * BK + acol];
        d[0] = pack_bf2(a0.x, a0.y); d[1] = pack_bf2(a0.z, a0.w);
        d[2] = pack_bf2(a1.x, a1.y); d[3] = pack_bf2(a1.z, a1.w);
    };
    auto storeW = [&](int buf) {
        unsigned* d = (unsigned*)&lB[buf][t * BK];
#pragma unroll
        for (int q = 0; q < 8; ++q) {
            d[2 * q]     = pack_bf2(rw[q].x, rw[q].y);
            d[2 * q + 1] = pack_bf2(rw[q].z, rw[q].w);
        }
    };

    // prologue: stage tile 0
    loadA(0); loadW(0);
    storeA(0); storeW(0);
    __syncthreads();

    const int nk = K / BK;
    for (int kt = 0; kt < nk; ++kt) {
        const int cur = kt & 1;
        const bool more = (kt + 1 < nk);
        if (more) { loadA((kt + 1) * BK); loadW((kt + 1) * BK); }  // overlap w/ WMMA

        BF16x16 af;
#pragma unroll
        for (int j = 0; j < 8; ++j) {
            int kk = hl * 16 + j * 2;
            af.u[j] = *(const unsigned*)&lA[cur][(wmg * 16 + l16) * BK + kk];
        }
#pragma unroll
        for (int nt = 0; nt < 8; ++nt) {
            BF16x16 bf;
#pragma unroll
            for (int j = 0; j < 8; ++j)
                bf.u[j] = *(const unsigned*)&lB[cur][(wng * 128 + nt * 16 + l16) * BK + hl * 16 + j * 2];
            acc[nt] = __builtin_amdgcn_wmma_f32_16x16x32_bf16(
                false, af.v, false, bf.v, (short)0, acc[nt], false, false);
        }
        if (more) { storeA(cur ^ 1); storeW(cur ^ 1); __syncthreads(); }
    }

#pragma unroll
    for (int nt = 0; nt < 8; ++nt) {
#pragma unroll
        for (int j = 0; j < 8; ++j) {
            int row = m0 + wmg * 16 + j + 8 * hl;
            int col = n0 + wng * 128 + nt * 16 + l16;
            float v = acc[nt][j];
            if (EPI == EPI_BIAS)           v += aux1[col];
            else if (EPI == EPI_BIAS_SILU) { v += aux1[col]; v = v / (1.f + __expf(-v)); }
            else if (EPI == EPI_BIAS_TEMB) v += aux1[col] + aux2[(size_t)(row / rpb) * N + col];
            else if (EPI == EPI_RESID)     v += aux1[(size_t)row * N + col];
            else if (EPI == EPI_GELU)      v = 0.5f * v * (1.f + erff(v * 0.70710678118f));
            C[(size_t)row * N + col] = v;
        }
    }
}

// ---------------------------------------------------------------------------
// Fused attention, one wave per (b,h).  S = (q sc)(k sc)^T -> fp32 softmax in
// LDS -> O = P@V, all via WMMA bf16.  593 keys padded to 608.  All gathers
// use clamped indices (unconditional loads); padding handled by value
// selects (score := -inf) and zeroed P columns.
// ---------------------------------------------------------------------------
__global__ __launch_bounds__(32) void attn_kernel(
    const float* __restrict__ qbuf,   // [B*16, 1024]
    const float* __restrict__ kv,     // [B*593, 2048]
    float* __restrict__ obuf)         // [B*16, 1024]
{
    constexpr int KL = 593, KLP = 608, DD = 1024, KVW = 2048;
    const int bh = blockIdx.x;
    const int b  = bh >> 4;
    const int h  = bh & 15;
    const int lane = threadIdx.x;
    const int l16 = lane & 15;
    const int hl  = lane >> 4;

    __shared__ float  sS[16 * KLP];
    __shared__ __bf16 sP[16 * KLP];

    BF16x16 aq[2];
    const float* qrow = qbuf + (size_t)(b * 16 + l16) * DD + h * 64;
#pragma unroll
    for (int ks = 0; ks < 2; ++ks)
#pragma unroll
        for (int j = 0; j < 8; ++j) {
            int kk = ks * 32 + hl * 16 + j * 2;
            float2 q2 = *(const float2*)(qrow + kk);
            aq[ks].u[j] = pack_bf2(q2.x, q2.y);
        }

    const float* Kbase = kv + (size_t)b * KL * KVW + h * 64;
    for (int kt = 0; kt < KLP / 16; ++kt) {
        v8f acc;
#pragma unroll
        for (int j = 0; j < 8; ++j) acc[j] = 0.f;
        const int klg = kt * 16 + l16;
        const bool ok = (klg < KL);
        const int klc = ok ? klg : (KL - 1);           // clamped: safe + finite
        const float* kr0 = Kbase + (size_t)klc * KVW;
#pragma unroll
        for (int ks = 0; ks < 2; ++ks) {
            BF16x16 bk;
#pragma unroll
            for (int j = 0; j < 8; ++j) {
                int dh = ks * 32 + hl * 16 + j * 2;
                float2 k2 = *(const float2*)(kr0 + dh);
                bk.u[j] = pack_bf2(k2.x, k2.y);
            }
            acc = __builtin_amdgcn_wmma_f32_16x16x32_bf16(
                false, aq[ks].v, false, bk.v, (short)0, acc, false, false);
        }
#pragma unroll
        for (int j = 0; j < 8; ++j)
            sS[(j + 8 * hl) * KLP + klg] = ok ? acc[j] * 0.125f : -3.0e38f;
    }
    __syncthreads();

    // softmax: 2 lanes per row (lane r and r+16 split even/odd columns)
    {
        const int r = l16;
        float mx = -3.0e38f;
        for (int c = hl; c < KL; c += 2) mx = fmaxf(mx, sS[r * KLP + c]);
        mx = fmaxf(mx, __shfl_xor(mx, 16, 32));
        float sum = 0.f;
        for (int c = hl; c < KL; c += 2) {
            float e = __expf(sS[r * KLP + c] - mx);
            sS[r * KLP + c] = e; sum += e;
        }
        sum += __shfl_xor(sum, 16, 32);
        const float invs = 1.f / sum;
        for (int c = hl; c < KLP; c += 2)
            sP[r * KLP + c] = (c < KL) ? (__bf16)(sS[r * KLP + c] * invs) : (__bf16)0.f;
    }
    __syncthreads();

    v8f oacc[4];
#pragma unroll
    for (int i = 0; i < 4; ++i)
#pragma unroll
        for (int j = 0; j < 8; ++j) oacc[i][j] = 0.f;

    const float* Vbase = kv + (size_t)b * KL * KVW + DD + h * 64;
    for (int ks = 0; ks < KLP / 32; ++ks) {
        BF16x16 ap;
#pragma unroll
        for (int j = 0; j < 8; ++j) {
            int kk = ks * 32 + hl * 16 + j * 2;
            ap.u[j] = *(const unsigned*)&sP[l16 * KLP + kk];
        }
#pragma unroll
        for (int nt = 0; nt < 4; ++nt) {
            BF16x16 bv;
#pragma unroll
            for (int j = 0; j < 8; ++j) {
                int kl  = ks * 32 + hl * 16 + j * 2;
                int k0c = (kl     < KL) ? kl     : (KL - 1);   // clamped: P=0 there
                int k1c = (kl + 1 < KL) ? kl + 1 : (KL - 1);
                float x0 = Vbase[(size_t)k0c * KVW + nt * 16 + l16];
                float x1 = Vbase[(size_t)k1c * KVW + nt * 16 + l16];
                bv.u[j] = pack_bf2(x0, x1);
            }
            oacc[nt] = __builtin_amdgcn_wmma_f32_16x16x32_bf16(
                false, ap.v, false, bv.v, (short)0, oacc[nt], false, false);
        }
    }
#pragma unroll
    for (int nt = 0; nt < 4; ++nt)
#pragma unroll
        for (int j = 0; j < 8; ++j)
            obuf[(size_t)(b * 16 + j + 8 * hl) * DD + h * 64 + nt * 16 + l16] = oacc[nt][j];
}

// per-row LayerNorm statistics: st[r] = {mean, rsqrt(var + eps)}
__global__ __launch_bounds__(256) void rowstats_kernel(
    const float* __restrict__ X, float2* __restrict__ st, int ncols)
{
    const int r = blockIdx.x;
    const int t = threadIdx.x;
    const float* x = X + (size_t)r * ncols;
    float s = 0.f, ss = 0.f;
    for (int c = t; c < ncols; c += 256) { float v = x[c]; s += v; ss += v * v; }
    __shared__ float red[256], red2[256];
    red[t] = s; red2[t] = ss;
    __syncthreads();
    for (int o = 128; o > 0; o >>= 1) {
        if (t < o) { red[t] += red[t + o]; red2[t] += red2[t + o]; }
        __syncthreads();
    }
    if (t == 0) {
        float mean = red[0] / ncols;
        float var  = red2[0] / ncols - mean * mean;
        st[r] = make_float2(mean, rsqrtf(var + 1e-5f));
    }
}

// fold AdaLN modulation into effective gain/bias: Gm = g*(1+sc), Bm = b*(1+sc)+sh
__global__ void modfold_kernel(
    const float* __restrict__ g, const float* __restrict__ bt,
    const float* __restrict__ mod, int shift_off, int scale_off,
    float* __restrict__ Gm, float* __restrict__ Bm)
{
    int i = blockIdx.x * 256 + threadIdx.x;   // over B*D
    int b = i >> 10, c = i & 1023;
    float sc = mod[(size_t)b * 4096 + scale_off + c];
    float sh = mod[(size_t)b * 4096 + shift_off + c];
    Gm[i] = g[c] * (1.f + sc);
    Bm[i] = bt[c] * (1.f + sc) + sh;
}

// plain LayerNorm (final output)
__global__ __launch_bounds__(256) void ln_kernel(
    const float* __restrict__ X, const float* __restrict__ g,
    const float* __restrict__ bt, float* __restrict__ out, int ncols)
{
    const int r = blockIdx.x;
    const int t = threadIdx.x;
    const float* x = X + (size_t)r * ncols;
    float s = 0.f, ss = 0.f;
    for (int c = t; c < ncols; c += 256) { float v = x[c]; s += v; ss += v * v; }
    __shared__ float red[256], red2[256];
    red[t] = s; red2[t] = ss;
    __syncthreads();
    for (int o = 128; o > 0; o >>= 1) {
        if (t < o) { red[t] += red[t + o]; red2[t] += red2[t + o]; }
        __syncthreads();
    }
    const float mean = red[0] / ncols;
    const float inv  = rsqrtf(red2[0] / ncols - mean * mean + 1e-5f);
    for (int c = t; c < ncols; c += 256)
        out[(size_t)r * ncols + c] = (x[c] - mean) * inv * g[c] + bt[c];
}

// diffusers Timesteps(320, flip_sin_to_cos=True): [cos | sin]
__global__ void sinus_kernel(const int* __restrict__ ts, float* __restrict__ emb)
{
    const int b = blockIdx.x;
    const int j = threadIdx.x;
    const float t = (float)ts[b];
    const int halfT = 160;
    const int i = (j < halfT) ? j : (j - halfT);
    const float f = __expf(-9.210340372f * (float)i / (float)halfT);
    emb[(size_t)b * 320 + j] = (j < halfT) ? __cosf(t * f) : __sinf(t * f);
}

__global__ void silu_kernel(const float* __restrict__ in, float* __restrict__ out, int n)
{
    int i = blockIdx.x * 256 + threadIdx.x;
    if (i < n) { float v = in[i]; out[i] = v / (1.f + __expf(-v)); }
}

__global__ void bcast_kernel(const float* __restrict__ src, float* __restrict__ dst,
                             int per, int total)
{
    int i = blockIdx.x * 256 + threadIdx.x;
    if (i < total) dst[i] = src[i % per];
}

// ---------------------------------------------------------------------------
extern "C" void kernel_launch(void* const* d_in, const int* in_sizes, int n_in,
                              void* d_out, int out_size, void* d_ws, size_t ws_size,
                              hipStream_t stream)
{
    (void)in_sizes; (void)n_in; (void)out_size; (void)ws_size;
    const float* x          = (const float*)d_in[0];
    const int*   timestep   = (const int*)  d_in[1];
    const float* latents    = (const float*)d_in[2];
    const float* proj_in_w  = (const float*)d_in[3];
    const float* proj_in_b  = (const float*)d_in[4];
    const float* time_w1    = (const float*)d_in[5];
    const float* time_b1    = (const float*)d_in[6];
    const float* time_w2    = (const float*)d_in[7];
    const float* time_b2    = (const float*)d_in[8];
    const float* ln1_g      = (const float*)d_in[9];
    const float* ln1_b      = (const float*)d_in[10];
    const float* ln2_g      = (const float*)d_in[11];
    const float* ln2_b      = (const float*)d_in[12];
    const float* q_w        = (const float*)d_in[13];
    const float* kv_w       = (const float*)d_in[14];
    const float* out_w      = (const float*)d_in[15];
    const float* ffln_g     = (const float*)d_in[16];
    const float* ffln_b     = (const float*)d_in[17];
    const float* ff_w1      = (const float*)d_in[18];
    const float* ff_w2      = (const float*)d_in[19];
    const float* ada_w      = (const float*)d_in[20];
    const float* ada_b      = (const float*)d_in[21];
    const float* proj_out_w = (const float*)d_in[22];
    const float* proj_out_b = (const float*)d_in[23];
    const float* nout_g     = (const float*)d_in[24];
    const float* nout_b     = (const float*)d_in[25];
    float* out = (float*)d_out;

    constexpr int Bb = 64, Nn = 577, Ee = 768, Dd = 1024, Ll = 8, Qq = 16,
                  Hh = 16, FFf = 4096, Tt = 320, Oo = 1024;
    constexpr int KL = Nn + Qq;  // 593

    float* p = (float*)d_ws;
    auto take = [&](size_t n) { float* r = p; p += n; return r; };
    float*  t_emb  = take((size_t)Bb * Tt);
    float*  act1   = take((size_t)Bb * Dd);
    float*  temb   = take((size_t)Bb * Dd);
    float*  st     = take((size_t)Bb * Dd);
    float*  modb   = take((size_t)Bb * 4 * Dd);
    float*  xp     = take((size_t)Bb * Nn * Dd);
    float*  kvb    = take((size_t)Bb * KL * 2 * Dd);
    float*  lat    = take((size_t)Bb * Qq * Dd);
    float*  qb     = take((size_t)Bb * Qq * Dd);
    float*  ob     = take((size_t)Bb * Qq * Dd);
    float*  h1     = take((size_t)Bb * Qq * FFf);
    float*  outb   = take((size_t)Bb * Qq * Oo);
    float2* st_xp  = (float2*)take((size_t)Bb * Nn * 2);
    float2* st_la  = (float2*)take((size_t)Bb * Qq * 2);
    float2* st_lb  = (float2*)take((size_t)Bb * Qq * 2);
    float*  GmA    = take((size_t)Bb * Dd);
    float*  BmA    = take((size_t)Bb * Dd);
    float*  GmB    = take((size_t)Bb * Dd);
    float*  BmB    = take((size_t)Bb * Dd);

    dim3 blk(256);
    const float2* nost = nullptr;

    // timestep embedding MLP
    sinus_kernel<<<Bb, Tt, 0, stream>>>(timestep, t_emb);
    gemm_bf16<EPI_BIAS_SILU, A_PLAIN><<<dim3(Dd / BN, Bb / BM), blk, 0, stream>>>(
        t_emb, time_w1, act1, Bb, Dd, Tt, time_b1, nullptr, 1,
        nullptr, nost, nost, nullptr, nullptr, nullptr, nullptr, 1, 0);
    gemm_bf16<EPI_BIAS, A_PLAIN><<<dim3(Dd / BN, Bb / BM), blk, 0, stream>>>(
        act1, time_w2, temb, Bb, Dd, Dd, time_b2, nullptr, 1,
        nullptr, nost, nost, nullptr, nullptr, nullptr, nullptr, 1, 0);
    silu_kernel<<<(Bb * Dd) / 256, blk, 0, stream>>>(temb, st, Bb * Dd);
    bcast_kernel<<<(Bb * Qq * Dd) / 256, blk, 0, stream>>>(latents, lat, Qq * Dd, Bb * Qq * Dd);

    // xp = x @ proj_in_w^T + b + temb[batch]
    gemm_bf16<EPI_BIAS_TEMB, A_PLAIN><<<dim3(Dd / BN, (Bb * Nn) / BM), blk, 0, stream>>>(
        x, proj_in_w, xp, Bb * Nn, Dd, Ee, proj_in_b, temb, Nn,
        nullptr, nost, nost, nullptr, nullptr, nullptr, nullptr, 1, 0);

    // xp LN statistics are layer-invariant: compute once
    rowstats_kernel<<<Bb * Nn, blk, 0, stream>>>(xp, st_xp, Dd);

    for (int i = 0; i < Ll; ++i) {
        // AdaLN: mod = silu(temb) @ ada_w^T + ada_b   [B, 4D]
        gemm_bf16<EPI_BIAS, A_PLAIN><<<dim3((4 * Dd) / BN, Bb / BM), blk, 0, stream>>>(
            st, ada_w + (size_t)i * 4 * Dd * Dd, modb, Bb, 4 * Dd, Dd,
            ada_b + (size_t)i * 4 * Dd, nullptr, 1,
            nullptr, nost, nost, nullptr, nullptr, nullptr, nullptr, 1, 0);
        rowstats_kernel<<<Bb * Qq, blk, 0, stream>>>(lat, st_la, Dd);
        modfold_kernel<<<(Bb * Dd) / 256, blk, 0, stream>>>(
            ln2_g + (size_t)i * Dd, ln2_b + (size_t)i * Dd, modb, 0, Dd, GmA, BmA);

        // kv = concat(LN(xp), modLN(lat)) @ kv_w^T   (dominant GEMM)
        gemm_bf16<EPI_NONE, A_LNCAT><<<dim3((2 * Dd) / BN, (Bb * KL) / BM), blk, 0, stream>>>(
            xp, kv_w + (size_t)i * 2 * Dd * Dd, kvb, Bb * KL, 2 * Dd, Dd,
            nullptr, nullptr, 1,
            lat, st_xp, st_la, ln1_g + (size_t)i * Dd, ln1_b + (size_t)i * Dd,
            GmA, BmA, KL, Nn);
        // q = modLN(lat) @ q_w^T
        gemm_bf16<EPI_NONE, A_LNCAT><<<dim3(Dd / BN, (Bb * Qq) / BM), blk, 0, stream>>>(
            nullptr, q_w + (size_t)i * Dd * Dd, qb, Bb * Qq, Dd, Dd,
            nullptr, nullptr, 1,
            lat, nost, st_la, nullptr, nullptr, GmA, BmA, Qq, 0);
        attn_kernel<<<Bb * Hh, 32, 0, stream>>>(qb, kvb, ob);
        // lat = o @ out_w^T + lat
        gemm_bf16<EPI_RESID, A_PLAIN><<<dim3(Dd / BN, (Bb * Qq) / BM), blk, 0, stream>>>(
            ob, out_w + (size_t)i * Dd * Dd, lat, Bb * Qq, Dd, Dd,
            lat, nullptr, 1,
            nullptr, nost, nost, nullptr, nullptr, nullptr, nullptr, 1, 0);

        // FFN: h1 = gelu(modLN(lat) @ ff_w1^T) ; lat += h1 @ ff_w2^T
        rowstats_kernel<<<Bb * Qq, blk, 0, stream>>>(lat, st_lb, Dd);
        modfold_kernel<<<(Bb * Dd) / 256, blk, 0, stream>>>(
            ffln_g + (size_t)i * Dd, ffln_b + (size_t)i * Dd, modb, 2 * Dd, 3 * Dd, GmB, BmB);
        gemm_bf16<EPI_GELU, A_LNCAT><<<dim3(FFf / BN, (Bb * Qq) / BM), blk, 0, stream>>>(
            nullptr, ff_w1 + (size_t)i * FFf * Dd, h1, Bb * Qq, FFf, Dd,
            nullptr, nullptr, 1,
            lat, nost, st_lb, nullptr, nullptr, GmB, BmB, Qq, 0);
        gemm_bf16<EPI_RESID, A_PLAIN><<<dim3(Dd / BN, (Bb * Qq) / BM), blk, 0, stream>>>(
            h1, ff_w2 + (size_t)i * Dd * FFf, lat, Bb * Qq, Dd, FFf,
            lat, nullptr, 1,
            nullptr, nost, nost, nullptr, nullptr, nullptr, nullptr, 1, 0);
    }

    // out = LN(lat @ proj_out_w^T + b)
    gemm_bf16<EPI_BIAS, A_PLAIN><<<dim3(Oo / BN, (Bb * Qq) / BM), blk, 0, stream>>>(
        lat, proj_out_w, outb, Bb * Qq, Oo, Dd, proj_out_b, nullptr, 1,
        nullptr, nost, nost, nullptr, nullptr, nullptr, nullptr, 1, 0);
    ln_kernel<<<Bb * Qq, blk, 0, stream>>>(outb, nout_g, nout_b, out, Oo);
}
